// MultiHeadAttention_71347996721787
// MI455X (gfx1250) — compile-verified, matrix operand order checked by reading, not run on hardware
//
#include <hip/hip_runtime.h>
#include <hip/hip_bf16.h>
#include <cstdint>
#include <cstddef>

// ---------------------------------------------------------------------------
// MHA for MI455X (gfx1250): bf16 WMMA (v_wmma_f32_16x16x32_bf16) everywhere,
// async global->LDS staging (ASYNCcnt) for bf16 tiles, global_prefetch hints,
// native v_cvt bf16 conversion.  B=4, S=2048, E=1024, H=16, Dh=64.
// ---------------------------------------------------------------------------

typedef unsigned short u16;
typedef __attribute__((ext_vector_type(4)))  int    v4i;
typedef __attribute__((ext_vector_type(8)))  u16    v8u;
typedef __attribute__((ext_vector_type(16))) u16    v16u;
typedef __attribute__((ext_vector_type(16))) __bf16 v16bf;
typedef __attribute__((ext_vector_type(8)))  float  v8f;

#define AS1 __attribute__((address_space(1)))
#define AS3 __attribute__((address_space(3)))

#define BB 4
#define SS 2048
#define EE 1024
#define HH 16
#define DH 64
#define MM (BB * SS)   // 8192 rows

// float -> bf16 via native convert (v_cvt_bf16_f32 / packed pair form)
static __device__ inline u16 f2bf(float f) {
  union { __bf16 h; u16 u; } x;
  x.h = (__bf16)f;
  return x.u;
}

static __device__ inline v16u cat8(v8u lo, v8u hi) {
  return __builtin_shufflevector(lo, hi, 0,1,2,3,4,5,6,7,8,9,10,11,12,13,14,15);
}

static __device__ inline v8f wmma_bf16(v16u a, v16u b, v8f c) {
  union { v16u u; v16bf bf; } A, B;
  A.u = a; B.u = b;
  return __builtin_amdgcn_wmma_f32_16x16x32_bf16(
      /*neg_a=*/false, A.bf, /*neg_b=*/false, B.bf,
      /*c_mod=*/(short)0, c, /*reuse_a=*/false, /*reuse_b=*/false);
}

static __device__ inline void st4bf(u16* d, float4 t) {
  d[0] = f2bf(t.x); d[1] = f2bf(t.y); d[2] = f2bf(t.z); d[3] = f2bf(t.w);
}

// Build a 16x32 (bf16) A/B fragment from a row of 16B-aligned u16 LDS/global.
// lane<16: K in {0..7, 16..23}; lane>=16: K in {8..15, 24..31}.
static __device__ inline v16u frag_row(const u16* p, int kb) {
  v8u lo = *(const v8u*)(p + kb);
  v8u hi = *(const v8u*)(p + kb + 16);
  return cat8(lo, hi);
}

// --- CDNA5 async global->LDS (16B per lane), tracked by ASYNCcnt -----------
// Builtin signature (from hipcc diagnostic): (v4i AS1*, v4i AS3*, i32, i32)
static __device__ inline void async_ld_b128(const u16* g, u16* l) {
#if __has_builtin(__builtin_amdgcn_global_load_async_to_lds_b128)
  __builtin_amdgcn_global_load_async_to_lds_b128(
      (AS1 v4i*)const_cast<u16*>(g), (AS3 v4i*)l, 0, 0);
#else
  unsigned loff = (unsigned)(uintptr_t)(AS3 u16*)l;       // 32-bit LDS offset
  unsigned long long ga = (unsigned long long)(uintptr_t)g;
  asm volatile("global_load_async_to_lds_b128 %0, %1, off"
               :: "v"(loff), "v"(ga) : "memory");
#endif
}

static __device__ inline void wait_async() {
#if __has_builtin(__builtin_amdgcn_s_wait_asynccnt)
  __builtin_amdgcn_s_wait_asynccnt(0);
#else
  asm volatile("s_wait_asynccnt 0x0" ::: "memory");
#endif
}

// ---------------------------------------------------------------------------
// Kernel 1: Y = X @ W^T + b, fp32 inputs, bf16 output scattered to [B,H,S,Dh].
// Block tile 128x128, K-step 32, 256 threads = 8 waves (4x2 wave grid).
// (fp32->bf16 conversion forces the VGPR staging path; add prefetch instead.)
// ---------------------------------------------------------------------------
__global__ __launch_bounds__(256) void k_proj(const float* __restrict__ X,
                                              const float* __restrict__ W,
                                              const float* __restrict__ bias,
                                              u16* __restrict__ Out) {
  __shared__ u16 As[128][40];   // [m][k], pitch 40 (80B, 16B-aligned)
  __shared__ u16 Bs[128][40];   // [n][k]

  const int tid  = threadIdx.x;
  const int lane = tid & 31;
  const int wv   = tid >> 5;
  const int wm   = wv & 3;          // 4 M sub-blocks of 32
  const int wn   = wv >> 2;         // 2 N sub-blocks of 64
  const int m0   = blockIdx.y * 128;
  const int n0   = blockIdx.x * 128;
  const int lrow = tid >> 1;        // 0..127 tile row to stage
  const int lseg = (tid & 1) << 4;  // 0 or 16 (k segment)
  const int colL = lane & 15;
  const int kb   = (lane < 16) ? 0 : 8;
  const int hi8  = (lane >> 4) << 3;

  v8f acc[2][4];
#pragma unroll
  for (int i = 0; i < 2; ++i)
#pragma unroll
    for (int j = 0; j < 4; ++j) acc[i][j] = v8f{};

  for (int k0 = 0; k0 < EE; k0 += 32) {
    {  // stage A tile (fp32 -> bf16)
      const float4* src = (const float4*)(X + (size_t)(m0 + lrow) * EE + k0 + lseg);
      float4 t0 = src[0], t1 = src[1], t2 = src[2], t3 = src[3];
      u16* dst = &As[lrow][lseg];
      st4bf(dst + 0, t0); st4bf(dst + 4, t1); st4bf(dst + 8, t2); st4bf(dst + 12, t3);
    }
    {  // stage B tile = W rows (fp32 -> bf16)
      const float4* src = (const float4*)(W + (size_t)(n0 + lrow) * EE + k0 + lseg);
      float4 t0 = src[0], t1 = src[1], t2 = src[2], t3 = src[3];
      u16* dst = &Bs[lrow][lseg];
      st4bf(dst + 0, t0); st4bf(dst + 4, t1); st4bf(dst + 8, t2); st4bf(dst + 12, t3);
    }
    if (k0 + 32 < EE) {  // prefetch next k-step tile rows
      __builtin_prefetch(X + (size_t)(m0 + lrow) * EE + k0 + 32 + lseg, 0, 1);
      __builtin_prefetch(W + (size_t)(n0 + lrow) * EE + k0 + 32 + lseg, 0, 1);
    }
    __syncthreads();

    v16u a[2], b[4];
#pragma unroll
    for (int mt = 0; mt < 2; ++mt)
      a[mt] = frag_row(&As[wm * 32 + mt * 16 + colL][0], kb);
#pragma unroll
    for (int nt = 0; nt < 4; ++nt)
      b[nt] = frag_row(&Bs[wn * 64 + nt * 16 + colL][0], kb);
#pragma unroll
    for (int mt = 0; mt < 2; ++mt)
#pragma unroll
      for (int nt = 0; nt < 4; ++nt)
        acc[mt][nt] = wmma_bf16(a[mt], b[nt], acc[mt][nt]);
    __syncthreads();
  }

  // Epilogue: + bias, bf16 scatter into [B,H,S,Dh]
#pragma unroll
  for (int nt = 0; nt < 4; ++nt) {
    const int n = n0 + wn * 64 + nt * 16 + colL;
    const float bv = bias[n];
    const int h = n >> 6, d = n & 63;
#pragma unroll
    for (int mt = 0; mt < 2; ++mt) {
#pragma unroll
      for (int r = 0; r < 8; ++r) {
        const int m  = m0 + wm * 32 + mt * 16 + r + hi8;
        const int bb = m >> 11, s = m & (SS - 1);
        Out[(((size_t)(bb * HH + h)) * SS + s) * DH + d] = f2bf(acc[mt][nt][r] + bv);
      }
    }
  }
}

// ---------------------------------------------------------------------------
// Kernel 2: flash attention. One workgroup = one (b,h) x 128 q-rows.
// 8 waves x 16 q rows. Keys in chunks of 64; online softmax.
// K tile staged with async global->LDS; V transposed through registers.
// ---------------------------------------------------------------------------
__global__ __launch_bounds__(256) void k_attn(const u16* __restrict__ Qh,
                                              const u16* __restrict__ Kh,
                                              const u16* __restrict__ Vh,
                                              const int* __restrict__ mask,
                                              u16* __restrict__ Cc) {
  __shared__ u16 Ks[64][72];       // [key][d]
  __shared__ u16 Vs[64][72];       // transposed: [d][key]
  __shared__ u16 Ps[8][16][72];    // per-wave P tile (16 q rows x 64 keys)

  const int tid  = threadIdx.x;
  const int lane = tid & 31;
  const int wv   = tid >> 5;
  const int bh   = blockIdx.y;                 // b*H + h
  const int q0   = blockIdx.x * 128 + wv * 16; // first q row of this wave
  const size_t headBase = (size_t)bh * SS * DH;

  const int colL = lane & 15;
  const int kb   = (lane < 16) ? 0 : 8;
  const int hi8  = (lane >> 4) << 3;

  // Q fragments (A-matrix, K = Dh split 0..31 / 32..63), loaded once
  v16u qa[2];
  {
    const u16* qp = Qh + headBase + (size_t)(q0 + colL) * DH;
    qa[0] = frag_row(qp, kb);
    qa[1] = frag_row(qp + 32, kb);
  }

  v8f o[4];
#pragma unroll
  for (int t = 0; t < 4; ++t) o[t] = v8f{};
  float ms[8], ls[8];
#pragma unroll
  for (int r = 0; r < 8; ++r) { ms[r] = -3.0e38f; ls[r] = 0.0f; }

  const float scale = 0.125f;           // 1/sqrt(64)
  const float L2E   = 1.44269504088896f;

  const int ldr  = tid >> 2;            // key row to stage (0..63)
  const int dseg = (tid & 3) << 4;      // d segment (0,16,32,48)

  for (int kt = 0; kt < SS / 64; ++kt) {
    const int key0 = kt * 64;
    {  // stage K tile [key][d]: async global->LDS, no VGPR round-trip
      const u16* src = Kh + headBase + (size_t)(key0 + ldr) * DH + dseg;
      async_ld_b128(src,     &Ks[ldr][dseg]);
      async_ld_b128(src + 8, &Ks[ldr][dseg + 8]);
    }
    {  // stage V tile transposed [d][key] (register transpose)
      const u16* src = Vh + headBase + (size_t)(key0 + ldr) * DH + dseg;
      v8u a0 = *(const v8u*)src, a1 = *(const v8u*)(src + 8);
#pragma unroll
      for (int i = 0; i < 8; ++i) Vs[dseg + i][ldr] = a0[i];
#pragma unroll
      for (int i = 0; i < 8; ++i) Vs[dseg + 8 + i][ldr] = a1[i];
    }
    if (kt + 1 < SS / 64) {  // prefetch next K/V tile rows
      __builtin_prefetch(Kh + headBase + (size_t)(key0 + 64 + ldr) * DH + dseg, 0, 1);
      __builtin_prefetch(Vh + headBase + (size_t)(key0 + 64 + ldr) * DH + dseg, 0, 1);
    }
    wait_async();
    __syncthreads();

    // scores S = Q . K^T (B-fragment rows come straight from K rows)
    v8f sc[4];
#pragma unroll
    for (int nt = 0; nt < 4; ++nt) {
      const u16* kp = &Ks[nt * 16 + colL][0];
      v8f c = v8f{};
      c = wmma_bf16(qa[0], frag_row(kp, kb), c);
      c = wmma_bf16(qa[1], frag_row(kp + 32, kb), c);
      sc[nt] = c;
    }

    // scale + mask
#pragma unroll
    for (int nt = 0; nt < 4; ++nt) {
      const int kcol = key0 + nt * 16 + colL;
#pragma unroll
      for (int r = 0; r < 8; ++r) {
        const int q = q0 + r + hi8;
        const int mv = mask[(size_t)q * SS + kcol];
        const float s = sc[nt][r] * scale;
        sc[nt][r] = (mv == 0) ? -1.0e9f : s;
      }
    }

    // online softmax (row reductions across each 16-lane half)
#pragma unroll
    for (int r = 0; r < 8; ++r) {
      float mx = fmaxf(fmaxf(sc[0][r], sc[1][r]), fmaxf(sc[2][r], sc[3][r]));
#pragma unroll
      for (int d = 1; d < 16; d <<= 1) mx = fmaxf(mx, __shfl_xor(mx, d, 32));
      const float mnew = fmaxf(ms[r], mx);
      const float resc = exp2f((ms[r] - mnew) * L2E);
      ms[r] = mnew;
      float rs = 0.0f;
#pragma unroll
      for (int nt = 0; nt < 4; ++nt) {
        const float p = exp2f((sc[nt][r] - mnew) * L2E);
        sc[nt][r] = p;
        rs += p;
      }
#pragma unroll
      for (int d = 1; d < 16; d <<= 1) rs += __shfl_xor(rs, d, 32);
      ls[r] = ls[r] * resc + rs;
#pragma unroll
      for (int dt = 0; dt < 4; ++dt) o[dt][r] *= resc;
    }

    // P (C-layout) -> bf16 in per-wave LDS, reload in A-fragment layout
#pragma unroll
    for (int nt = 0; nt < 4; ++nt)
#pragma unroll
      for (int r = 0; r < 8; ++r)
        Ps[wv][r + hi8][nt * 16 + colL] = f2bf(sc[nt][r]);

    v16u pa[2];
    {
      const u16* pp = &Ps[wv][colL][0];
      pa[0] = frag_row(pp, kb);        // keys 0..31 of chunk
      pa[1] = frag_row(pp + 32, kb);   // keys 32..63
    }

    // O += P . V   (B-fragment rows from transposed V)
#pragma unroll
    for (int dt = 0; dt < 4; ++dt) {
      const u16* vp = &Vs[dt * 16 + colL][0];
      o[dt] = wmma_bf16(pa[0], frag_row(vp, kb), o[dt]);
      o[dt] = wmma_bf16(pa[1], frag_row(vp + 32, kb), o[dt]);
    }
    __syncthreads();
  }

  // normalize + store bf16 into concat layout [B, S, E] with e = h*64 + d
  const int b_ = bh >> 4, h = bh & 15;
#pragma unroll
  for (int r = 0; r < 8; ++r) {
    const float inv = 1.0f / ls[r];
    const int q = q0 + r + hi8;
    const size_t rowbase = ((size_t)(b_ * SS + q)) * EE + h * DH;
#pragma unroll
    for (int dt = 0; dt < 4; ++dt)
      Cc[rowbase + dt * 16 + colL] = f2bf(o[dt][r] * inv);
  }
}

// ---------------------------------------------------------------------------
// Kernel 3: out = concat(bf16) @ Wo^T + bo, fp32 output.
// A tile (already bf16) staged via async global->LDS.
// ---------------------------------------------------------------------------
__global__ __launch_bounds__(256) void k_out(const u16* __restrict__ A,
                                             const float* __restrict__ W,
                                             const float* __restrict__ bias,
                                             float* __restrict__ Y) {
  __shared__ u16 As[128][40];
  __shared__ u16 Bs[128][40];

  const int tid  = threadIdx.x;
  const int lane = tid & 31;
  const int wv   = tid >> 5;
  const int wm   = wv & 3;
  const int wn   = wv >> 2;
  const int m0   = blockIdx.y * 128;
  const int n0   = blockIdx.x * 128;
  const int lrow = tid >> 1;
  const int lseg = (tid & 1) << 4;
  const int colL = lane & 15;
  const int kb   = (lane < 16) ? 0 : 8;
  const int hi8  = (lane >> 4) << 3;

  v8f acc[2][4];
#pragma unroll
  for (int i = 0; i < 2; ++i)
#pragma unroll
    for (int j = 0; j < 4; ++j) acc[i][j] = v8f{};

  for (int k0 = 0; k0 < EE; k0 += 32) {
    {  // stage A tile (bf16): async global->LDS
      const u16* src = A + (size_t)(m0 + lrow) * EE + k0 + lseg;
      async_ld_b128(src,     &As[lrow][lseg]);
      async_ld_b128(src + 8, &As[lrow][lseg + 8]);
    }
    {  // stage B tile = Wo rows (fp32 -> bf16)
      const float4* src = (const float4*)(W + (size_t)(n0 + lrow) * EE + k0 + lseg);
      float4 t0 = src[0], t1 = src[1], t2 = src[2], t3 = src[3];
      u16* dst = &Bs[lrow][lseg];
      st4bf(dst + 0, t0); st4bf(dst + 4, t1); st4bf(dst + 8, t2); st4bf(dst + 12, t3);
    }
    if (k0 + 32 < EE) {
      __builtin_prefetch(A + (size_t)(m0 + lrow) * EE + k0 + 32 + lseg, 0, 1);
      __builtin_prefetch(W + (size_t)(n0 + lrow) * EE + k0 + 32 + lseg, 0, 1);
    }
    wait_async();
    __syncthreads();

    v16u a[2], b[4];
#pragma unroll
    for (int mt = 0; mt < 2; ++mt)
      a[mt] = frag_row(&As[wm * 32 + mt * 16 + colL][0], kb);
#pragma unroll
    for (int nt = 0; nt < 4; ++nt)
      b[nt] = frag_row(&Bs[wn * 64 + nt * 16 + colL][0], kb);
#pragma unroll
    for (int mt = 0; mt < 2; ++mt)
#pragma unroll
      for (int nt = 0; nt < 4; ++nt)
        acc[mt][nt] = wmma_bf16(a[mt], b[nt], acc[mt][nt]);
    __syncthreads();
  }

#pragma unroll
  for (int nt = 0; nt < 4; ++nt) {
    const int n = n0 + wn * 64 + nt * 16 + colL;
    const float bv = bias[n];
#pragma unroll
    for (int mt = 0; mt < 2; ++mt) {
#pragma unroll
      for (int r = 0; r < 8; ++r) {
        const int m = m0 + wm * 32 + mt * 16 + r + hi8;
        Y[(size_t)m * EE + n] = acc[mt][nt][r] + bv;
      }
    }
  }
}

// ---------------------------------------------------------------------------
extern "C" void kernel_launch(void* const* d_in, const int* in_sizes, int n_in,
                              void* d_out, int out_size, void* d_ws, size_t ws_size,
                              hipStream_t stream) {
  (void)in_sizes; (void)n_in; (void)out_size; (void)ws_size;

  const float* q    = (const float*)d_in[0];
  const float* k    = (const float*)d_in[1];
  const float* v    = (const float*)d_in[2];
  const int*   mask = (const int*)d_in[3];
  const float* Wq   = (const float*)d_in[4];
  const float* bq   = (const float*)d_in[5];
  const float* Wk   = (const float*)d_in[6];
  const float* bk   = (const float*)d_in[7];
  const float* Wv   = (const float*)d_in[8];
  const float* bv   = (const float*)d_in[9];
  const float* Wo   = (const float*)d_in[10];
  const float* bo   = (const float*)d_in[11];

  // workspace: Qh | Kh | Vh | concat, each 8192x1024 bf16 (16 MB) = 64 MB
  u16* Qh = (u16*)d_ws;
  u16* Kh = Qh + (size_t)MM * EE;
  u16* Vh = Kh + (size_t)MM * EE;
  u16* Cc = Vh + (size_t)MM * EE;

  dim3 blk(256);
  dim3 gGemm(EE / 128, MM / 128);      // (8, 64)
  dim3 gAttn(SS / 128, BB * HH);       // (16, 64)

  k_proj<<<gGemm, blk, 0, stream>>>(q, Wq, bq, Qh);
  k_proj<<<gGemm, blk, 0, stream>>>(k, Wk, bk, Kh);
  k_proj<<<gGemm, blk, 0, stream>>>(v, Wv, bv, Vh);
  k_attn<<<gAttn, blk, 0, stream>>>(Qh, Kh, Vh, mask, Cc);
  k_out<<<gGemm, blk, 0, stream>>>(Cc, Wo, bo, (float*)d_out);
}
